// MultiViewFeatureExtractor_55619826483355
// MI455X (gfx1250) — matrix operand order repaired and checked by reading, not compiled, not run on hardware
//
#include <hip/hip_runtime.h>
#include <hip/hip_bf16.h>

// ---------- types ----------
typedef __bf16 bf16_t;
typedef __attribute__((ext_vector_type(16))) bf16_t v16bf;
typedef __attribute__((ext_vector_type(4)))  bf16_t bf16x4;
typedef __attribute__((ext_vector_type(8)))  float  v8f;
typedef __attribute__((ext_vector_type(4)))  float  f32x4;
typedef __attribute__((ext_vector_type(4)))  unsigned int u32x4;
typedef __attribute__((ext_vector_type(2)))  unsigned int u32x2;

union Frag { v16bf v; u32x4 u[2]; };

#define FLAG_BINDIAG 1
#define FLAG_RELU    2
#define FLAG_BF16OUT 4

// =====================================================================
// Generic 128x128-tile GEMM:  out = epilogue( A[rows,K] @ B_bf16[K,cols] )
// A is f32 (optionally binarized with unit diagonal -> A_hat of GCN).
// B is bf16 row-major. v_wmma_f32_16x16x32_bf16, double-buffered LDS,
// register-staged global loads (software pipeline, 1 barrier / K-step).
// Block: 256 threads = 8 waves; wave w computes rows [w*16,w*16+16) x 128 cols.
// =====================================================================
template <int FLAGS>
__global__ void __launch_bounds__(256)
wmma_gemm_kernel(const float* __restrict__ A, const bf16_t* __restrict__ B,
                 const float* __restrict__ bias, const float* __restrict__ rowscale,
                 void* __restrict__ out,
                 int K, int lda, int ldb, int ldo,
                 long long strideA, long long strideB, long long strideBias,
                 long long strideScale, long long strideOut)
{
    __shared__ bf16_t Atile[2][128][40];   // 128 rows x 32 k (+pad) x2 buffers
    __shared__ bf16_t Btile[2][128][40];   // 128 cols x 32 k (transposed) x2

    const int tid  = threadIdx.x;
    const int lane = tid & 31;
    const int wv   = tid >> 5;
    const int rowBlock = blockIdx.x * 128;
    const int colBase  = blockIdx.y * 128;
    const int z = blockIdx.z;

    A += (size_t)z * strideA;
    B += (size_t)z * strideB;
    if (bias)     bias     += (size_t)z * strideBias;
    if (rowscale) rowscale += (size_t)z * strideScale;
    float*  outf = (float*)out  + (size_t)z * strideOut;
    bf16_t* outh = (bf16_t*)out + (size_t)z * strideOut;

    f32x4 aR[4];   // staged A granules (f32x4 each)
    u32x2 bR[4];   // staged B granules (4 bf16 each)

    auto loadA = [&](int kb) {
        #pragma unroll
        for (int i = 0; i < 4; ++i) {
            int idx = tid + i * 256;            // 0..1023 float4 granules
            int r   = idx >> 3;                 // 0..127
            int kq  = idx & 7;                  // k quad
            const float* src = A + (size_t)(rowBlock + r) * lda + kb + kq * 4;
            aR[i] = *reinterpret_cast<const f32x4*>(src);
            __builtin_prefetch(src + 32, 0, 1); // next K tile (speculative)
        }
    };
    auto storeA = [&](int kb, int p) {
        #pragma unroll
        for (int i = 0; i < 4; ++i) {
            int idx = tid + i * 256;
            int r   = idx >> 3;
            int kq  = idx & 7;
            f32x4 f = aR[i];
            if (FLAGS & FLAG_BINDIAG) {
                int gr = rowBlock + r, gk = kb + kq * 4;
                f.x = (f.x != 0.f) ? 1.f : 0.f;
                f.y = (f.y != 0.f) ? 1.f : 0.f;
                f.z = (f.z != 0.f) ? 1.f : 0.f;
                f.w = (f.w != 0.f) ? 1.f : 0.f;
                if (gr == gk)     f.x = 1.f;    // A_hat = A + I
                if (gr == gk + 1) f.y = 1.f;
                if (gr == gk + 2) f.z = 1.f;
                if (gr == gk + 3) f.w = 1.f;
            }
            bf16x4 h = { (bf16_t)f.x, (bf16_t)f.y, (bf16_t)f.z, (bf16_t)f.w };
            *reinterpret_cast<bf16x4*>(&Atile[p][r][kq * 4]) = h;
        }
    };
    auto loadB = [&](int kb) {
        #pragma unroll
        for (int i = 0; i < 4; ++i) {
            int idx = tid + i * 256;            // 0..1023 (4-bf16 granules)
            int kk  = idx >> 5;                 // 0..31
            int cq  = idx & 31;                 // col quad
            bR[i] = *reinterpret_cast<const u32x2*>(
                        B + (size_t)(kb + kk) * ldb + colBase + cq * 4);
        }
    };
    auto storeB = [&](int p) {
        #pragma unroll
        for (int i = 0; i < 4; ++i) {
            int idx = tid + i * 256;
            int kk  = idx >> 5;
            int cq  = idx & 31;
            union { u32x2 u; bf16_t h[4]; } w;
            w.u = bR[i];
            Btile[p][cq * 4 + 0][kk] = w.h[0];
            Btile[p][cq * 4 + 1][kk] = w.h[1];
            Btile[p][cq * 4 + 2][kk] = w.h[2];
            Btile[p][cq * 4 + 3][kk] = w.h[3];
        }
    };

    v8f acc[8] = {};

    loadA(0);
    loadB(0);
    int p = 0;
    for (int kb = 0; kb < K; kb += 32) {
        storeA(kb, p);
        storeB(p);
        __syncthreads();
        if (kb + 32 < K) {          // issue next tile's global loads early
            loadA(kb + 32);
            loadB(kb + 32);
        }
        // A fragment: 16x32 bf16 per ISA layout
        // lane<16: row=lane, K=0..7 (v0-3) + 16..23 (v4-7); lane>=16: K=8..15 + 24..31
        Frag a;
        {
            int r    = wv * 16 + (lane & 15);
            int koff = (lane < 16) ? 0 : 8;
            a.u[0] = *reinterpret_cast<const u32x4*>(&Atile[p][r][koff]);
            a.u[1] = *reinterpret_cast<const u32x4*>(&Atile[p][r][koff + 16]);
        }
        #pragma unroll
        for (int n = 0; n < 8; ++n) {
            // B fragment 32x16: lane<16 holds K=0..7 + 16..23 of its col; lane>=16 K=8..15 + 24..31
            Frag b;
            int c   = n * 16 + (lane & 15);
            int kk0 = (lane < 16) ? 0 : 8;
            b.u[0] = *reinterpret_cast<const u32x4*>(&Btile[p][c][kk0]);
            b.u[1] = *reinterpret_cast<const u32x4*>(&Btile[p][c][kk0 + 16]);
            acc[n] = __builtin_amdgcn_wmma_f32_16x16x32_bf16(
                         false, a.v, false, b.v, (short)0, acc[n], false, false);
        }
        p ^= 1;
    }

    // epilogue: C layout: VGPR r, lane l -> M=(l<16? r : 8+r), N=l&15
    #pragma unroll
    for (int n = 0; n < 8; ++n) {
        int col = colBase + n * 16 + (lane & 15);
        float bcol = bias ? bias[col] : 0.f;
        #pragma unroll
        for (int r = 0; r < 8; ++r) {
            int row = rowBlock + wv * 16 + ((lane < 16) ? r : 8 + r);
            float vr = acc[n][r];
            if (rowscale) vr *= rowscale[row];
            vr += bcol;
            if (FLAGS & FLAG_RELU) vr = fmaxf(vr, 0.f);
            if (FLAGS & FLAG_BF16OUT) outh[(size_t)row * ldo + col] = (bf16_t)vr;
            else                      outf[(size_t)row * ldo + col] = vr;
        }
    }
}

// =====================================================================
// dinv[v*N+i] = rsqrt(1 + #nonzeros in adj row)   (one wave per row)
// =====================================================================
__global__ void __launch_bounds__(256)
deg_dinv_kernel(const float* __restrict__ adjs, float* __restrict__ dinv)
{
    int lane = threadIdx.x & 31, wv = threadIdx.x >> 5;
    int row = blockIdx.x * 8 + wv;            // 0 .. V*N-1
    const float* a = adjs + (size_t)row * 4096;
    float s = 0.f;
    for (int k = lane * 4; k < 4096; k += 128) {
        f32x4 f = *reinterpret_cast<const f32x4*>(a + k);
        s += ((f.x != 0.f) ? 1.f : 0.f) + ((f.y != 0.f) ? 1.f : 0.f)
           + ((f.z != 0.f) ? 1.f : 0.f) + ((f.w != 0.f) ? 1.f : 0.f);
    }
    for (int off = 16; off > 0; off >>= 1) s += __shfl_down(s, off, 32);
    if (lane == 0) dinv[row] = rsqrtf(1.f + s);
}

// y = bf16(x * rowscale[i/rowlen])  (rowscale may be null)
__global__ void __launch_bounds__(256)
f32_to_bf16_scale(const float* __restrict__ x, const float* __restrict__ rowscale,
                  bf16_t* __restrict__ y, int rowlen, int count)
{
    int i = blockIdx.x * 256 + threadIdx.x;
    if (i >= count) return;
    float s = rowscale ? rowscale[i / rowlen] : 1.f;
    y[i] = (bf16_t)(x[i] * s);
}

// summaries: mean over N of h2  -> summ[V*H]  (one wave per column)
__global__ void __launch_bounds__(256)
colmean_kernel(const float* __restrict__ h2, float* __restrict__ summ)
{
    int lane = threadIdx.x & 31, wv = threadIdx.x >> 5;
    int col = blockIdx.x * 8 + wv;            // 0..511
    int v = col >> 7, h = col & 127;
    const float* p = h2 + ((size_t)v * 4096) * 128 + h;
    float s = 0.f;
    for (int n = lane; n < 4096; n += 32) s += p[(size_t)n * 128];
    for (int off = 16; off > 0; off >>= 1) s += __shfl_down(s, off, 32);
    if (lane == 0) summ[col] = s * (1.f / 4096.f);
}

// attention: scores = tanh(summ@wa1+ba1)@wa2+ba2; softmax over V=4
__global__ void __launch_bounds__(256)
attn_kernel(const float* __restrict__ summ, const float* __restrict__ wa1,
            const float* __restrict__ ba1, const float* __restrict__ wa2,
            const float* __restrict__ ba2, float* __restrict__ attn_ws,
            float* __restrict__ attn_out)
{
    __shared__ float part[256];
    __shared__ float sc[4];
    int t = threadIdx.x, v = t >> 6, j = t & 63;
    float a = ba1[j];
    for (int h = 0; h < 128; ++h) a += summ[v * 128 + h] * wa1[h * 64 + j];
    part[t] = tanhf(a) * wa2[j];
    __syncthreads();
    if (t < 4) {
        float s = ba2[0];
        for (int jj = 0; jj < 64; ++jj) s += part[t * 64 + jj];
        sc[t] = s;
    }
    __syncthreads();
    if (t == 0) {
        float m = fmaxf(fmaxf(sc[0], sc[1]), fmaxf(sc[2], sc[3]));
        float e0 = expf(sc[0] - m), e1 = expf(sc[1] - m);
        float e2 = expf(sc[2] - m), e3 = expf(sc[3] - m);
        float inv = 1.f / (e0 + e1 + e2 + e3);
        attn_ws[0] = e0 * inv; attn_ws[1] = e1 * inv;
        attn_ws[2] = e2 * inv; attn_ws[3] = e3 * inv;
        attn_out[0] = e0 * inv; attn_out[1] = e1 * inv;
        attn_out[2] = e2 * inv; attn_out[3] = e3 * inv;
    }
}

// fin[n, v*128+h] = attn[v] * h2[v][n][h]   (f32, feeds fusion GEMM as A)
__global__ void __launch_bounds__(256)
make_fin_kernel(const float* __restrict__ h2, const float* __restrict__ attn,
                float* __restrict__ fin)
{
    int i = blockIdx.x * 256 + threadIdx.x;   // < 4096*512
    int n = i >> 9, c = i & 511, v = c >> 7, h = c & 127;
    fin[i] = attn[v] * h2[(((size_t)v * 4096) + n) * 128 + h];
}

// =====================================================================
static void launch_gemm(dim3 grid, hipStream_t stream,
                        const float* A, const bf16_t* B, const float* bias,
                        const float* rowscale, void* out,
                        int K, int lda, int ldb, int ldo,
                        long long sA, long long sB, long long sBias,
                        long long sScale, long long sOut, int flags)
{
    dim3 blk(256);
    switch (flags) {
    case FLAG_BINDIAG | FLAG_RELU:
        wmma_gemm_kernel<FLAG_BINDIAG | FLAG_RELU><<<grid, blk, 0, stream>>>(
            A, B, bias, rowscale, out, K, lda, ldb, ldo, sA, sB, sBias, sScale, sOut);
        break;
    case FLAG_BF16OUT:
        wmma_gemm_kernel<FLAG_BF16OUT><<<grid, blk, 0, stream>>>(
            A, B, bias, rowscale, out, K, lda, ldb, ldo, sA, sB, sBias, sScale, sOut);
        break;
    case FLAG_RELU:
        wmma_gemm_kernel<FLAG_RELU><<<grid, blk, 0, stream>>>(
            A, B, bias, rowscale, out, K, lda, ldb, ldo, sA, sB, sBias, sScale, sOut);
        break;
    default:
        wmma_gemm_kernel<0><<<grid, blk, 0, stream>>>(
            A, B, bias, rowscale, out, K, lda, ldb, ldo, sA, sB, sBias, sScale, sOut);
        break;
    }
}

extern "C" void kernel_launch(void* const* d_in, const int* in_sizes, int n_in,
                              void* d_out, int out_size, void* d_ws, size_t ws_size,
                              hipStream_t stream)
{
    (void)in_sizes; (void)n_in; (void)out_size; (void)ws_size;
    const int V = 4, N = 4096, H = 128;

    const float* adjs = (const float*)d_in[0];
    const float* w1   = (const float*)d_in[1];
    const float* b1   = (const float*)d_in[2];
    const float* w2   = (const float*)d_in[3];
    const float* b2   = (const float*)d_in[4];
    const float* wa1  = (const float*)d_in[5];
    const float* ba1  = (const float*)d_in[6];
    const float* wa2  = (const float*)d_in[7];
    const float* ba2  = (const float*)d_in[8];
    const float* wf1  = (const float*)d_in[9];
    const float* bf1  = (const float*)d_in[10];
    const float* wf2  = (const float*)d_in[11];
    const float* bf2  = (const float*)d_in[12];

    float* out       = (float*)d_out;
    float* fused_out = out;                     // [N,128]
    float* attn_out  = out + (size_t)N * 128;   // [V]
    float* h2_out    = attn_out + V;            // [V,N,H]

    // workspace layout (slots aliased by liveness)
    char* ws = (char*)d_ws;
    const size_t OFF_DINV  = 0;            // 64 KB   V*N f32
    const size_t OFF_SLOTA = 65536;        // 8 MB    h1 (f32), later fin (f32)
    const size_t OFF_SLOTB = 8454144;      // 4 MB    Bs1 (bf16), later G1 (f32)
    const size_t OFF_SLOTC = 12648448;     // 4 MB    Bs2 (bf16)
    const size_t OFF_W2BF  = 16842752;     // 128 KB
    const size_t OFF_WF1BF = 16973824;     // 256 KB
    const size_t OFF_WF2BF = 17235968;     // 64 KB
    const size_t OFF_SUMM  = 17301504;     // 2 KB
    const size_t OFF_ATTN  = 17303552;     // 16 B

    float*  dinv    = (float*)(ws + OFF_DINV);
    float*  h1      = (float*)(ws + OFF_SLOTA);
    float*  fin     = (float*)(ws + OFF_SLOTA);
    bf16_t* Bs1     = (bf16_t*)(ws + OFF_SLOTB);
    float*  G1      = (float*)(ws + OFF_SLOTB);
    bf16_t* Bs2     = (bf16_t*)(ws + OFF_SLOTC);
    bf16_t* w2bf    = (bf16_t*)(ws + OFF_W2BF);
    bf16_t* wf1bf   = (bf16_t*)(ws + OFF_WF1BF);
    bf16_t* wf2bf   = (bf16_t*)(ws + OFF_WF2BF);
    float*  summ    = (float*)(ws + OFF_SUMM);
    float*  attn_ws = (float*)(ws + OFF_ATTN);

    // 1) degrees -> dinv
    deg_dinv_kernel<<<V * N / 8, 256, 0, stream>>>(adjs, dinv);

    // 2) prescale / convert weights to bf16
    f32_to_bf16_scale<<<(V * N * H + 255) / 256, 256, 0, stream>>>(w1, dinv, Bs1, H, V * N * H);
    f32_to_bf16_scale<<<(V * H * H + 255) / 256, 256, 0, stream>>>(w2, nullptr, w2bf, H, V * H * H);
    f32_to_bf16_scale<<<(512 * 256 + 255) / 256, 256, 0, stream>>>(wf1, nullptr, wf1bf, 256, 512 * 256);
    f32_to_bf16_scale<<<(256 * 128 + 255) / 256, 256, 0, stream>>>(wf2, nullptr, wf2bf, 128, 256 * 128);

    dim3 gV(N / 128, 1, V);

    // 3) h1 = relu(dinv .* (A_hat @ (dinv.*w1)) + b1)
    launch_gemm(gV, stream, adjs, Bs1, b1, dinv, h1, N, N, H, H,
                (long long)N * N, (long long)N * H, H, N, (long long)N * H,
                FLAG_BINDIAG | FLAG_RELU);

    // 4) Bs2 = bf16(dinv .* (h1 @ w2))
    launch_gemm(gV, stream, h1, w2bf, nullptr, dinv, Bs2, H, H, H, H,
                (long long)N * H, (long long)H * H, 0, N, (long long)N * H,
                FLAG_BF16OUT);

    // 5) h2 = relu(dinv .* (A_hat @ Bs2) + b2)  -> straight into d_out
    launch_gemm(gV, stream, adjs, Bs2, b2, dinv, h2_out, N, N, H, H,
                (long long)N * N, (long long)N * H, H, N, (long long)N * H,
                FLAG_BINDIAG | FLAG_RELU);

    // 6) summaries + attention softmax
    colmean_kernel<<<V * H / 8, 256, 0, stream>>>(h2_out, summ);
    attn_kernel<<<1, 256, 0, stream>>>(summ, wa1, ba1, wa2, ba2, attn_ws, attn_out);

    // 7) fusion input: fin[n, v*H+h] = attn[v]*h2[v][n][h]
    make_fin_kernel<<<(N * V * H) / 256, 256, 0, stream>>>(h2_out, attn_ws, fin);

    // 8) G1 = relu(fin @ wf1 + bf1)   [N,512]@[512,256]
    dim3 gF1(N / 128, 2, 1);
    launch_gemm(gF1, stream, fin, wf1bf, bf1, nullptr, G1, 512, 512, 256, 256,
                0, 0, 0, 0, 0, FLAG_RELU);

    // 9) fused = G1 @ wf2 + bf2       [N,256]@[256,128] -> d_out
    dim3 gF2(N / 128, 1, 1);
    launch_gemm(gF2, stream, G1, wf2bf, bf2, nullptr, fused_out, 256, 256, 128, 128,
                0, 0, 0, 0, 0, 0);
}
